// EdgeTypeRGCN_28707561406844
// MI455X (gfx1250) — matrix operand order repaired and checked by reading, not compiled, not run on hardware
//
#include <hip/hip_runtime.h>

typedef __attribute__((ext_vector_type(16))) _Float16 v16h;
typedef __attribute__((ext_vector_type(8)))  _Float16 v8h;
typedef __attribute__((ext_vector_type(4)))  _Float16 v4h;
typedef __attribute__((ext_vector_type(8)))  float    v8f;
typedef __attribute__((ext_vector_type(4)))  float    v4f;

#define NNODES 100000
#define NEDGES 1600000
#define RREL   16
#define NBASES 4
#define CHUNK  2048
#define FEAT   64

__device__ inline v16h cat16(v8h lo, v8h hi) {
  return __builtin_shufflevector(lo, hi, 0,1,2,3,4,5,6,7,8,9,10,11,12,13,14,15);
}

__device__ inline v16h loadB16(const _Float16* p) {
  v8h lo = *(const v8h*)p;
  v8h hi = *(const v8h*)(p + 8);
  return cat16(lo, hi);
}

// A-fragment for one 16-row x 32-K slab from an f16 row-major 64-wide matrix:
// lane half 'grp' supplies K in [grp*16, grp*16+16).
__device__ inline v16h loadA16(const _Float16* row, int grp) {
  v8h lo = *(const v8h*)(row + grp * 16);
  v8h hi = *(const v8h*)(row + grp * 16 + 8);
  return cat16(lo, hi);
}

__device__ inline v8f wmma_f16(v16h a, v16h b, v8f c) {
  return __builtin_amdgcn_wmma_f32_16x16x32_f16(false, a, false, b, (short)0, c,
                                                false, false);
}

// ---------------------------------------------------------------------------
// Kernel 1: W_r = sum_b comp[r,b] * bases[b]  -> Wt[r][n][k] (transposed, f16)
//           loopT[n][k] = loop_weight[k][n]   (transposed, f16)
// ---------------------------------------------------------------------------
__global__ __launch_bounds__(256) void rgcn_prep(
    const float* __restrict__ bases, const float* __restrict__ comp,
    const float* __restrict__ loopw, _Float16* __restrict__ Wt,
    _Float16* __restrict__ loopT) {
  int t = blockIdx.x * 256 + threadIdx.x;
  if (t < RREL * FEAT * FEAT) {
    int r = t >> 12;
    int n = (t >> 6) & 63;
    int k = t & 63;
    float acc = 0.f;
#pragma unroll
    for (int b = 0; b < NBASES; ++b)
      acc += comp[r * NBASES + b] * bases[(b * FEAT + k) * FEAT + n];
    Wt[t] = (_Float16)acc;
  } else {
    int t2 = t - RREL * FEAT * FEAT;
    if (t2 < FEAT * FEAT) {
      int n = t2 >> 6;
      int k = t2 & 63;
      loopT[t2] = (_Float16)loopw[k * FEAT + n];
    }
  }
}

// ---------------------------------------------------------------------------
// Kernel 1b: one-time fp32 -> f16 cast of node features (12.8MB, L2-resident;
// every feature row is gathered ~16x afterwards, so this halves gather bytes
// and removes all v_cvt from the edge hot loop).
// ---------------------------------------------------------------------------
__global__ __launch_bounds__(256) void rgcn_cast(
    const float* __restrict__ x, _Float16* __restrict__ xh) {
  size_t t4 = ((size_t)blockIdx.x * 256 + threadIdx.x) * 4;
  if (t4 < (size_t)NNODES * FEAT) {
    v4f a = *(const v4f*)(x + t4);
    v4h h;
    h[0] = (_Float16)a[0]; h[1] = (_Float16)a[1];
    h[2] = (_Float16)a[2]; h[3] = (_Float16)a[3];
    *(v4h*)(xh + t4) = h;
  }
}

// ---------------------------------------------------------------------------
// Kernel 2: fused message passing. Per 2048-edge chunk: bin by relation in
// LDS, then each wave owns 2 relations, keeps B=W_r in regs, and runs
// 16-edge x 64x64 f16 WMMA groups, scattering with hw f32 atomics into agg
// (L2-resident, 25.6MB).
// ---------------------------------------------------------------------------
__global__ __launch_bounds__(256) void rgcn_edges(
    const _Float16* __restrict__ xh, const _Float16* __restrict__ Wt,
    const int* __restrict__ src, const int* __restrict__ dst,
    const int* __restrict__ etype, float* __restrict__ agg) {
  __shared__ int s_src[CHUNK];
  __shared__ int s_dst[CHUNK];
  __shared__ int s_cnt[RREL];
  __shared__ int s_off[RREL + 1];
  __shared__ int s_cur[RREL];

  const int tid  = threadIdx.x;
  const int base = blockIdx.x * CHUNK;

  if (tid < RREL) s_cnt[tid] = 0;
  __syncthreads();
  for (int i = tid; i < CHUNK; i += 256) {
    int e = base + i;
    if (e < NEDGES) atomicAdd(&s_cnt[etype[e]], 1);
  }
  __syncthreads();
  if (tid == 0) {
    int acc = 0;
    for (int r = 0; r < RREL; ++r) {
      s_off[r] = acc;
      s_cur[r] = acc;
      acc += s_cnt[r];
    }
    s_off[RREL] = acc;
  }
  __syncthreads();
  for (int i = tid; i < CHUNK; i += 256) {
    int e = base + i;
    if (e < NEDGES) {
      int r    = etype[e];
      int slot = atomicAdd(&s_cur[r], 1);
      s_src[slot] = src[e];
      s_dst[slot] = dst[e];
    }
  }
  __syncthreads();

  const int wave = tid >> 5;
  const int lane = tid & 31;
  const int grp  = lane >> 4;  // 0: lanes 0-15, 1: lanes 16-31
  const int ln   = lane & 15;

  for (int rr = 0; rr < 2; ++rr) {
    const int rel = wave * 2 + rr;
    const int beg = s_off[rel], end = s_off[rel + 1];
    if (beg >= end) continue;

    // B = W_rel, kept in 64 VGPRs, reused for all this relation's edges.
    v16h Bm[4][2];
    const _Float16* wb = Wt + (size_t)rel * FEAT * FEAT;
#pragma unroll
    for (int nt = 0; nt < 4; ++nt) {
      const _Float16* rowp = wb + (size_t)(nt * 16 + ln) * FEAT + grp * 16;
      Bm[nt][0] = loadB16(rowp);
      Bm[nt][1] = loadB16(rowp + 32);
    }

    for (int gs = beg; gs < end; gs += 16) {
      const int len = min(16, end - gs);
      const int ei  = gs + min(ln, len - 1);  // clamp pad rows
      const int sv  = s_src[ei];
      const int dv  = s_dst[ei];

      const _Float16* xp = xh + (size_t)sv * FEAT;
      v16h A0 = loadA16(xp, grp);
      v16h A1 = loadA16(xp + 32, grp);

      v8f c0 = {}, c1 = {}, c2 = {}, c3 = {};
      c0 = wmma_f16(A0, Bm[0][0], c0); c0 = wmma_f16(A1, Bm[0][1], c0);
      c1 = wmma_f16(A0, Bm[1][0], c1); c1 = wmma_f16(A1, Bm[1][1], c1);
      c2 = wmma_f16(A0, Bm[2][0], c2); c2 = wmma_f16(A1, Bm[2][1], c2);
      c3 = wmma_f16(A0, Bm[3][0], c3); c3 = wmma_f16(A1, Bm[3][1], c3);

      // Broadcast all 8 dst ids first so the bpermutes pipeline under one
      // wait, then issue the atomic clauses.
      int dd[8];
#pragma unroll
      for (int v = 0; v < 8; ++v) dd[v] = __shfl(dv, v + grp * 8, 32);

      if (len == 16) {  // fast path: no per-row EXEC masking
#pragma unroll
        for (int v = 0; v < 8; ++v) {
          float* ap = agg + (size_t)dd[v] * FEAT + ln;
          unsafeAtomicAdd(ap + 0,  c0[v]);
          unsafeAtomicAdd(ap + 16, c1[v]);
          unsafeAtomicAdd(ap + 32, c2[v]);
          unsafeAtomicAdd(ap + 48, c3[v]);
        }
      } else {
#pragma unroll
        for (int v = 0; v < 8; ++v) {
          if (v + grp * 8 < len) {
            float* ap = agg + (size_t)dd[v] * FEAT + ln;
            unsafeAtomicAdd(ap + 0,  c0[v]);
            unsafeAtomicAdd(ap + 16, c1[v]);
            unsafeAtomicAdd(ap + 32, c2[v]);
            unsafeAtomicAdd(ap + 48, c3[v]);
          }
        }
      }
    }
  }
}

// ---------------------------------------------------------------------------
// Kernel 3: out = LayerNorm(LeakyReLU(agg + x @ loop_weight + bias)) * gamma
//           + beta. One wave per 16-node tile; WMMA for the self-loop GEMM,
//           row stats via shfl_xor over 16-lane halves (C rows live per-half).
// ---------------------------------------------------------------------------
__global__ __launch_bounds__(256) void rgcn_finish(
    const _Float16* __restrict__ xh, const _Float16* __restrict__ loopT,
    const float* __restrict__ agg, const float* __restrict__ bias,
    const float* __restrict__ gamma, const float* __restrict__ beta,
    float* __restrict__ out) {
  const int tid  = threadIdx.x;
  const int wave = tid >> 5;
  const int lane = tid & 31;
  const int grp  = lane >> 4;
  const int ln   = lane & 15;
  const int tile = blockIdx.x * 8 + wave;
  const int nb   = tile * 16;
  if (nb >= NNODES) return;  // wave-uniform

  v16h Bm[4][2];
#pragma unroll
  for (int nt = 0; nt < 4; ++nt) {
    const _Float16* rowp = loopT + (size_t)(nt * 16 + ln) * FEAT + grp * 16;
    Bm[nt][0] = loadB16(rowp);
    Bm[nt][1] = loadB16(rowp + 32);
  }

  const int nodeA = min(nb + ln, NNODES - 1);
  const _Float16* xp = xh + (size_t)nodeA * FEAT;
  v16h A0 = loadA16(xp, grp);
  v16h A1 = loadA16(xp + 32, grp);

  v8f c[4];
#pragma unroll
  for (int nt = 0; nt < 4; ++nt) {
    float bv = bias[nt * 16 + ln];
#pragma unroll
    for (int v = 0; v < 8; ++v) {
      int node = min(nb + v + grp * 8, NNODES - 1);
      c[nt][v] = agg[(size_t)node * FEAT + nt * 16 + ln] + bv;
    }
  }
#pragma unroll
  for (int nt = 0; nt < 4; ++nt) {
    c[nt] = wmma_f16(A0, Bm[nt][0], c[nt]);
    c[nt] = wmma_f16(A1, Bm[nt][1], c[nt]);
  }

  // LeakyReLU(0.1)
#pragma unroll
  for (int nt = 0; nt < 4; ++nt)
#pragma unroll
    for (int v = 0; v < 8; ++v) {
      float o = c[nt][v];
      c[nt][v] = o >= 0.f ? o : 0.1f * o;
    }

  // Row mean (rows v+8*grp live in vgpr v across the 16 lanes of each half).
  float mu[8], rs[8];
#pragma unroll
  for (int v = 0; v < 8; ++v) {
    float s = c[0][v] + c[1][v] + c[2][v] + c[3][v];
    for (int m = 1; m < 16; m <<= 1) s += __shfl_xor(s, m, 32);
    mu[v] = s * (1.0f / 64.0f);
  }
#pragma unroll
  for (int v = 0; v < 8; ++v) {
    float d0 = c[0][v] - mu[v], d1 = c[1][v] - mu[v];
    float d2 = c[2][v] - mu[v], d3 = c[3][v] - mu[v];
    float s = d0 * d0 + d1 * d1 + d2 * d2 + d3 * d3;
    for (int m = 1; m < 16; m <<= 1) s += __shfl_xor(s, m, 32);
    rs[v] = rsqrtf(s * (1.0f / 64.0f) + 1e-5f);
  }

#pragma unroll
  for (int nt = 0; nt < 4; ++nt) {
    float g  = gamma[nt * 16 + ln];
    float bt = beta[nt * 16 + ln];
#pragma unroll
    for (int v = 0; v < 8; ++v) {
      int node = nb + v + grp * 8;
      if (node < NNODES)
        out[(size_t)node * FEAT + nt * 16 + ln] =
            (c[nt][v] - mu[v]) * rs[v] * g + bt;
    }
  }
}

// ---------------------------------------------------------------------------
extern "C" void kernel_launch(void* const* d_in, const int* in_sizes, int n_in,
                              void* d_out, int out_size, void* d_ws,
                              size_t ws_size, hipStream_t stream) {
  const float* node_feat = (const float*)d_in[0];
  const float* bases     = (const float*)d_in[1];
  const float* comp      = (const float*)d_in[2];
  const float* loopw     = (const float*)d_in[3];
  const float* bias      = (const float*)d_in[4];
  const float* gamma     = (const float*)d_in[5];
  const float* beta      = (const float*)d_in[6];
  const int*   src       = (const int*)d_in[7];
  const int*   dst       = (const int*)d_in[8];
  const int*   et        = (const int*)d_in[9];

  // Workspace layout (all 16B-aligned):
  //   agg   : N*64 f32   (25.6 MB)
  //   Wt    : R*64*64 f16 (128 KB)
  //   loopT : 64*64 f16   (8 KB)
  //   xh    : N*64 f16   (12.8 MB)
  float*    agg   = (float*)d_ws;
  _Float16* Wt    = (_Float16*)((char*)d_ws + (size_t)NNODES * FEAT * 4);
  _Float16* loopT = Wt + RREL * FEAT * FEAT;
  _Float16* xh    = loopT + FEAT * FEAT;

  hipMemsetAsync(agg, 0, (size_t)NNODES * FEAT * sizeof(float), stream);

  rgcn_prep<<<(RREL * FEAT * FEAT + FEAT * FEAT + 255) / 256, 256, 0, stream>>>(
      bases, comp, loopw, Wt, loopT);

  rgcn_cast<<<(NNODES * FEAT / 4 + 255) / 256, 256, 0, stream>>>(node_feat, xh);

  rgcn_edges<<<(NEDGES + CHUNK - 1) / CHUNK, 256, 0, stream>>>(
      xh, Wt, src, dst, et, agg);

  int tiles = (NNODES + 15) / 16;
  rgcn_finish<<<(tiles + 7) / 8, 256, 0, stream>>>(xh, loopT, agg, bias, gamma,
                                                   beta, (float*)d_out);
}
